// GroupedQueryAttention_48902497632772
// MI455X (gfx1250) — matrix operand order repaired and checked by reading, not compiled, not run on hardware
//
#include <hip/hip_runtime.h>

// ---------------- problem constants (match reference) ----------------
#define B_   2
#define T_   2048
#define D_   2048
#define H_   32
#define KV_  8
#define HD_  64
#define GROUPS_ 4
#define SCALE_ 0.125f            // 1/sqrt(64)
#define EPS_   1e-6f
#define LNFREQ_ 0.28782313662425572f   // ln(10000)/32

typedef __bf16 bf16;
typedef __attribute__((ext_vector_type(16))) __bf16 v16bf;
typedef __attribute__((ext_vector_type(8)))  __bf16 v8bf;
typedef __attribute__((ext_vector_type(4)))  __bf16 v4bf;
typedef __attribute__((ext_vector_type(8)))  float  v8f;
typedef __attribute__((ext_vector_type(4)))  int    v4i_;

// ---- CDNA5 async global->LDS copy (ASYNCcnt path), with safe fallback ----
// Probe-derived signature: (v4i addrspace(1)* gsrc, v4i addrspace(3)* lds, imm, imm)
#if __has_builtin(__builtin_amdgcn_global_load_async_to_lds_b128)
#define HAVE_ASYNC_LDS 1
#define CPASYNC16(l, g)                                                        \
  __builtin_amdgcn_global_load_async_to_lds_b128(                              \
      (__attribute__((address_space(1))) v4i_*)(g),                            \
      (__attribute__((address_space(3))) v4i_*)(l), 0, 0)
#if __has_builtin(__builtin_amdgcn_s_wait_asynccnt)
#define ASYNC_WAIT() __builtin_amdgcn_s_wait_asynccnt(0)
#else
#define ASYNC_WAIT() asm volatile("s_wait_asynccnt 0" ::: "memory")
#endif
#else
#define CPASYNC16(l, g) (*(v8bf*)(l) = *(const v8bf*)(g))
#define ASYNC_WAIT() ((void)0)
#endif

// float -> bf16, round-to-nearest-even (bit ops)
static __device__ __forceinline__ bf16 f2b(float f) {
  union { float f; unsigned u; } c; c.f = f;
  unsigned r = c.u + 0x7FFFu + ((c.u >> 16) & 1u);
  union { unsigned short s; bf16 b; } o; o.s = (unsigned short)(r >> 16);
  return o.b;
}

static __device__ __forceinline__ v16bf cat16(v8bf lo, v8bf hi) {
  return __builtin_shufflevector(lo, hi, 0,1,2,3,4,5,6,7,8,9,10,11,12,13,14,15);
}
static __device__ __forceinline__ v8bf ld8(const bf16* p) { return *(const v8bf*)p; }

static __device__ __forceinline__ v8f wmma_bf16(v16bf a, v16bf b, v8f c) {
  return __builtin_amdgcn_wmma_f32_16x16x32_bf16(false, a, false, b, (short)0, c, false, false);
}

// ---------------- kernel 1: f32 -> bf16 convert (x) ----------------
__global__ __launch_bounds__(256) void cvt_f32_bf16(const float* __restrict__ in,
                                                    bf16* __restrict__ outp) {
  size_t i = ((size_t)blockIdx.x * 256 + threadIdx.x) * 4;
  float4 f = *(const float4*)(in + i);
  v4bf o = { f2b(f.x), f2b(f.y), f2b(f.z), f2b(f.w) };
  *(v4bf*)(outp + i) = o;
}

// ---------------- kernel 2: LDS-tiled transpose + bf16 convert ----------------
// W: [K][N] f32 row-major -> Wt: [N][K] bf16 row-major
__global__ __launch_bounds__(256) void transpose_bf16_k(const float* __restrict__ W,
                                                        bf16* __restrict__ Wt,
                                                        int K, int N) {
  __shared__ float tile[32][33];
  int n0 = blockIdx.x * 32, k0 = blockIdx.y * 32;
  int tx = threadIdx.x & 31, ty = threadIdx.x >> 5;
#pragma unroll
  for (int j = 0; j < 32; j += 8)
    tile[ty + j][tx] = W[(size_t)(k0 + ty + j) * N + n0 + tx];
  __syncthreads();
#pragma unroll
  for (int j = 0; j < 32; j += 8)
    Wt[(size_t)(n0 + ty + j) * K + k0 + tx] = f2b(tile[tx][ty + j]);
}

// ---------------- kernel 3: bf16 WMMA GEMM with fused epilogues ----------------
// A[M][K] bf16 row-major, Bt[N][K] bf16 (pre-transposed B).
// block = 128 (4 waves); each wave owns a 16x64 C tile == one full head (HD=64).
// MODE 0: C f32 plain store           (output projection)
// MODE 1: rmsnorm(qs)+RoPE+SCALE -> bf16 Cb[row][col]      (Q)
// MODE 2: rmsnorm(ks)+RoPE       -> bf16 Cb[row][col]      (K)
// MODE 3: bf16, transposed store -> Cb[b][kv][hd][t]       (V^T)
template <int MODE>
__global__ __launch_bounds__(128) void gemm_bf16_wmma(const bf16* __restrict__ A,
                                                      const bf16* __restrict__ Bt,
                                                      float* __restrict__ Cf,
                                                      bf16* __restrict__ Cb,
                                                      const float* __restrict__ scale,
                                                      int N, int K) {
  const int lane = threadIdx.x & 31;
  const int wave = threadIdx.x >> 5;
  const int half = lane >> 4;
  const int l16  = lane & 15;
  const int m0 = blockIdx.x * 16;
  const int n0 = blockIdx.y * 256 + wave * 64;

  v8f acc0 = {}, acc1 = {}, acc2 = {}, acc3 = {};
  const int klo = half * 16;
  const int kk  = half * 16;
  const bf16* arow = A + (size_t)(m0 + l16) * K;
  const bf16* b0p  = Bt + (size_t)(n0 +  0 + l16) * K + kk;
  const bf16* b1p  = Bt + (size_t)(n0 + 16 + l16) * K + kk;
  const bf16* b2p  = Bt + (size_t)(n0 + 32 + l16) * K + kk;
  const bf16* b3p  = Bt + (size_t)(n0 + 48 + l16) * K + kk;

  for (int k = 0; k < K; k += 32) {
    __builtin_prefetch(arow + k + 256, 0, 1);   // global_prefetch_b8
    v16bf a  = cat16(ld8(arow + k + klo), ld8(arow + k + klo + 8));
    v16bf b0 = cat16(ld8(b0p + k), ld8(b0p + k + 8));
    v16bf b1 = cat16(ld8(b1p + k), ld8(b1p + k + 8));
    v16bf b2 = cat16(ld8(b2p + k), ld8(b2p + k + 8));
    v16bf b3 = cat16(ld8(b3p + k), ld8(b3p + k + 8));
    acc0 = wmma_bf16(a, b0, acc0);
    acc1 = wmma_bf16(a, b1, acc1);
    acc2 = wmma_bf16(a, b2, acc2);
    acc3 = wmma_bf16(a, b3, acc3);
  }

  if constexpr (MODE == 0) {
#pragma unroll
    for (int v = 0; v < 8; ++v) {
      size_t r = (size_t)(m0 + half * 8 + v) * N + n0 + l16;
      Cf[r +  0] = acc0[v];
      Cf[r + 16] = acc1[v];
      Cf[r + 32] = acc2[v];
      Cf[r + 48] = acc3[v];
    }
  } else if constexpr (MODE == 3) {
    // V: bf16 convert + transpose to [b][kv][hd][t]
    bf16* base = Cb + (size_t)(n0 >> 6) * (size_t)HD_ * T_;
#pragma unroll
    for (int v = 0; v < 8; ++v) {
      int row = m0 + half * 8 + v;
      int t = row & (T_ - 1);
      bf16* vb = base + (size_t)(row >> 11) * KV_ * HD_ * T_;
      vb[(size_t)( 0 + l16) * T_ + t] = f2b(acc0[v]);
      vb[(size_t)(16 + l16) * T_ + t] = f2b(acc1[v]);
      vb[(size_t)(32 + l16) * T_ + t] = f2b(acc2[v]);
      vb[(size_t)(48 + l16) * T_ + t] = f2b(acc3[v]);
    }
  } else {
    // Q/K: fused RMSNorm (over the wave's 64-wide head) + RoPE (+ SCALE for Q)
    const float sc0 = scale[ 0 + l16], sc1 = scale[16 + l16];
    const float sc2 = scale[32 + l16], sc3 = scale[48 + l16];
    const float f0 = __expf((float)l16        * -LNFREQ_);   // inv_freq[l16]
    const float f1 = __expf((float)(16 + l16) * -LNFREQ_);   // inv_freq[16+l16]
#pragma unroll
    for (int v = 0; v < 8; ++v) {
      float a0 = acc0[v], a1 = acc1[v], a2 = acc2[v], a3 = acc3[v];
      float ss = a0 * a0 + a1 * a1 + a2 * a2 + a3 * a3;
      ss += __shfl_xor(ss, 1);  ss += __shfl_xor(ss, 2);
      ss += __shfl_xor(ss, 4);  ss += __shfl_xor(ss, 8);
      float inv = rsqrtf(ss * (1.0f / 64.0f) + EPS_);
      a0 *= inv * sc0;  a1 *= inv * sc1;  a2 *= inv * sc2;  a3 *= inv * sc3;
      int row = m0 + half * 8 + v;
      float t = (float)(row & (T_ - 1));
      float c0 = __cosf(t * f0), s0 = __sinf(t * f0);
      float c1 = __cosf(t * f1), s1 = __sinf(t * f1);
      float o0 = a0 * c0 - a2 * s0, o2 = a2 * c0 + a0 * s0;  // pair (hd, hd+32)
      float o1 = a1 * c1 - a3 * s1, o3 = a3 * c1 + a1 * s1;
      if constexpr (MODE == 1) { o0 *= SCALE_; o1 *= SCALE_; o2 *= SCALE_; o3 *= SCALE_; }
      bf16* orow = Cb + (size_t)row * N + n0 + l16;
      orow[ 0] = f2b(o0);
      orow[16] = f2b(o1);
      orow[32] = f2b(o2);
      orow[48] = f2b(o3);
    }
  }
}

// ---------------- kernel 4: causal GQA flash attention ----------------
// block = 128 (4 waves) = the 4 q-heads sharing one kv head; K/V chunks staged
// once per block into LDS with async global->LDS copies.
__global__ __launch_bounds__(128) void gqa_attn_wmma(const bf16* __restrict__ Qb,
                                                     const bf16* __restrict__ Kb,
                                                     const bf16* __restrict__ Vtb,
                                                     bf16* __restrict__ Ab) {
  __shared__ __align__(16) bf16 ldsK[32 * 64];        // [t_local][hd]
  __shared__ __align__(16) bf16 ldsV[64 * 32];        // [hd][t_local]
  __shared__ __align__(16) bf16 ldsP[4 * 16 * 32];    // per-wave P staging

  const int tid  = threadIdx.x;
  const int lane = tid & 31;
  const int w    = tid >> 5;
  const int half = lane >> 4;
  const int l16  = lane & 15;
  const int NQT  = T_ / 16;
  const int qt  = blockIdx.x % NQT;
  const int kvh = (blockIdx.x / NQT) % KV_;
  const int b   = blockIdx.x / (NQT * KV_);
  const int h   = kvh * GROUPS_ + w;
  const int t0  = qt * 16;

  // per-wave Q A-frags straight from global (natural [t][hd] layout)
  const bf16* qrow = Qb + ((size_t)(b * T_ + t0 + l16) * H_ + h) * HD_;
  const int klo = half * 16;
  const v16bf qa0 = cat16(ld8(qrow + klo),      ld8(qrow + klo + 8));
  const v16bf qa1 = cat16(ld8(qrow + 32 + klo), ld8(qrow + 32 + klo + 8));

  v8f o0 = {}, o1 = {}, o2 = {}, o3 = {};
  float mstat[8], lstat[8];
#pragma unroll
  for (int v = 0; v < 8; ++v) { mstat[v] = -__builtin_inff(); lstat[v] = 0.0f; }

  const size_t kRowBase = (size_t)b * T_ * (KV_ * HD_) + (size_t)kvh * HD_;
  const bf16* vbase = Vtb + (size_t)(b * KV_ + kvh) * HD_ * T_;
  bf16* ldsPw = ldsP + w * (16 * 32);
  const int kk = half * 16;

  for (int c = 0; c < t0 + 16; c += 32) {
    // ---- stage K (32x64) and V^T (64x32) chunks into LDS: 512 x 16B, 4/thread
    {
      int u0 = tid, u1 = tid + 128;
      int r = u0 >> 3, sg = u0 & 7;
      CPASYNC16(ldsK + r * 64 + sg * 8,
                Kb + kRowBase + (size_t)(c + r) * (KV_ * HD_) + sg * 8);
      r = u1 >> 3; sg = u1 & 7;
      CPASYNC16(ldsK + r * 64 + sg * 8,
                Kb + kRowBase + (size_t)(c + r) * (KV_ * HD_) + sg * 8);
      int hd = u0 >> 2; sg = u0 & 3;
      CPASYNC16(ldsV + hd * 32 + sg * 8, vbase + (size_t)hd * T_ + c + sg * 8);
      hd = u1 >> 2; sg = u1 & 3;
      CPASYNC16(ldsV + hd * 32 + sg * 8, vbase + (size_t)hd * T_ + c + sg * 8);
      ASYNC_WAIT();
      __syncthreads();
    }

    // ---- S = Q K^T (16 x 32): K rows from LDS are natural B-frags
    const bf16* kr0 = ldsK + l16 * 64;
    const bf16* kr1 = ldsK + (16 + l16) * 64;
    v16bf kb00 = cat16(ld8(kr0 + kk),      ld8(kr0 + kk + 8));
    v16bf kb01 = cat16(ld8(kr0 + 32 + kk), ld8(kr0 + 32 + kk + 8));
    v16bf kb10 = cat16(ld8(kr1 + kk),      ld8(kr1 + kk + 8));
    v16bf kb11 = cat16(ld8(kr1 + 32 + kk), ld8(kr1 + 32 + kk + 8));
    v8f s0 = {}, s1 = {};
    s0 = wmma_bf16(qa0, kb00, s0);  s0 = wmma_bf16(qa1, kb01, s0);
    s1 = wmma_bf16(qa0, kb10, s1);  s1 = wmma_bf16(qa1, kb11, s1);

    // ---- causal mask (only the diagonal-intersecting chunk)
    if (c + 31 > t0) {
#pragma unroll
      for (int v = 0; v < 8; ++v) {
        int tq = t0 + half * 8 + v;
        if (c + l16 > tq)      s0[v] = -__builtin_inff();
        if (c + 16 + l16 > tq) s1[v] = -__builtin_inff();
      }
    }

    // ---- online softmax; row stats reduced across each 16-lane half
    float alpha[8];
#pragma unroll
    for (int v = 0; v < 8; ++v) {
      float rmax = fmaxf(s0[v], s1[v]);
      rmax = fmaxf(rmax, __shfl_xor(rmax, 1));
      rmax = fmaxf(rmax, __shfl_xor(rmax, 2));
      rmax = fmaxf(rmax, __shfl_xor(rmax, 4));
      rmax = fmaxf(rmax, __shfl_xor(rmax, 8));
      float mnew = fmaxf(mstat[v], rmax);
      alpha[v] = __expf(mstat[v] - mnew);
      float p0 = __expf(s0[v] - mnew);
      float p1 = __expf(s1[v] - mnew);
      float rs = p0 + p1;
      rs += __shfl_xor(rs, 1);
      rs += __shfl_xor(rs, 2);
      rs += __shfl_xor(rs, 4);
      rs += __shfl_xor(rs, 8);
      lstat[v] = lstat[v] * alpha[v] + rs;
      mstat[v] = mnew;
      int m = half * 8 + v;                   // C-layout row
      ldsPw[m * 32 + l16]      = f2b(p0);
      ldsPw[m * 32 + 16 + l16] = f2b(p1);
    }
#pragma unroll
    for (int v = 0; v < 8; ++v) {
      o0[v] *= alpha[v]; o1[v] *= alpha[v]; o2[v] *= alpha[v]; o3[v] *= alpha[v];
    }
    // per-wave DS ops complete in order -> P store/load round trip is safe
    v16bf pa = cat16(*(const v8bf*)(ldsPw + l16 * 32 + klo),
                     *(const v8bf*)(ldsPw + l16 * 32 + klo + 8));
    // ---- O += P * V : V^T rows from LDS are natural B-frags
    const bf16* vr0 = ldsV + (size_t)( 0 + l16) * 32 + kk;
    const bf16* vr1 = ldsV + (size_t)(16 + l16) * 32 + kk;
    const bf16* vr2 = ldsV + (size_t)(32 + l16) * 32 + kk;
    const bf16* vr3 = ldsV + (size_t)(48 + l16) * 32 + kk;
    o0 = wmma_bf16(pa, cat16(ld8(vr0), ld8(vr0 + 8)), o0);
    o1 = wmma_bf16(pa, cat16(ld8(vr1), ld8(vr1 + 8)), o1);
    o2 = wmma_bf16(pa, cat16(ld8(vr2), ld8(vr2 + 8)), o2);
    o3 = wmma_bf16(pa, cat16(ld8(vr3), ld8(vr3 + 8)), o3);
    __syncthreads();   // all waves done reading ldsK/ldsV before next stage
  }

  // ---- normalize and emit bf16 rows for the output projection
#pragma unroll
  for (int v = 0; v < 8; ++v) {
    float rl = 1.0f / lstat[v];
    int t = t0 + half * 8 + v;
    bf16* orow = Ab + (size_t)(b * T_ + t) * (H_ * HD_) + h * HD_;
    orow[ 0 + l16] = f2b(o0[v] * rl);
    orow[16 + l16] = f2b(o1[v] * rl);
    orow[32 + l16] = f2b(o2[v] * rl);
    orow[48 + l16] = f2b(o3[v] * rl);
  }
}

// ---------------- host: workspace carve + launch chain ----------------
extern "C" void kernel_launch(void* const* d_in, const int* in_sizes, int n_in,
                              void* d_out, int out_size, void* d_ws, size_t ws_size,
                              hipStream_t stream) {
  (void)in_sizes; (void)n_in; (void)out_size; (void)ws_size;
  const float* x  = (const float*)d_in[0];
  const float* Wq = (const float*)d_in[1];
  const float* Wk = (const float*)d_in[2];
  const float* Wv = (const float*)d_in[3];
  const float* Wo = (const float*)d_in[4];
  const float* qs = (const float*)d_in[5];
  const float* ks = (const float*)d_in[6];
  float* out = (float*)d_out;

  const size_t M  = (size_t)B_ * T_;       // 4096 rows
  const int    NQ = H_ * HD_;              // 2048
  const int    NK = KV_ * HD_;             // 512

  char* w = (char*)d_ws;
  auto take = [&](size_t bytes) -> char* {
    char* p = w; w += (bytes + 255) & ~(size_t)255; return p;
  };
  bf16* xb  = (bf16*)take(M * D_ * 2);
  bf16* Wqt = (bf16*)take((size_t)NQ * D_ * 2);
  bf16* Wkt = (bf16*)take((size_t)NK * D_ * 2);
  bf16* Wvt = (bf16*)take((size_t)NK * D_ * 2);
  bf16* Wot = (bf16*)take((size_t)D_ * NQ * 2);
  bf16* Qb  = (bf16*)take(M * NQ * 2);
  bf16* Kb  = (bf16*)take(M * NK * 2);
  bf16* Vtb = (bf16*)take(M * NK * 2);
  bf16* Ab  = (bf16*)take(M * NQ * 2);

  // 1) x -> bf16
  cvt_f32_bf16<<<(unsigned)((M * D_) / 1024), 256, 0, stream>>>(x, xb);
  // 2) weights -> transposed bf16 ([N][K])
  transpose_bf16_k<<<dim3(NQ / 32, D_ / 32), 256, 0, stream>>>(Wq, Wqt, D_, NQ);
  transpose_bf16_k<<<dim3(NK / 32, D_ / 32), 256, 0, stream>>>(Wk, Wkt, D_, NK);
  transpose_bf16_k<<<dim3(NK / 32, D_ / 32), 256, 0, stream>>>(Wv, Wvt, D_, NK);
  transpose_bf16_k<<<dim3(D_ / 32, NQ / 32), 256, 0, stream>>>(Wo, Wot, NQ, D_);
  // 3) QKV projections with fused RMSNorm+RoPE / V-transpose epilogues (WMMA)
  gemm_bf16_wmma<1><<<dim3(M / 16, NQ / 256), 128, 0, stream>>>(xb, Wqt, nullptr, Qb, qs, NQ, D_);
  gemm_bf16_wmma<2><<<dim3(M / 16, NK / 256), 128, 0, stream>>>(xb, Wkt, nullptr, Kb, ks, NK, D_);
  gemm_bf16_wmma<3><<<dim3(M / 16, NK / 256), 128, 0, stream>>>(xb, Wvt, nullptr, Vtb, nullptr, NK, D_);
  // 4) causal GQA flash attention (WMMA + async LDS staging)
  gqa_attn_wmma<<<B_ * KV_ * (T_ / 16), 128, 0, stream>>>(Qb, Kb, Vtb, Ab);
  // 5) output projection (WMMA) -> f32 d_out
  gemm_bf16_wmma<0><<<dim3(M / 16, D_ / 256), 128, 0, stream>>>(Ab, Wot, out, nullptr, nullptr, D_, NQ);
}
